// LSTMLearnerLayer_70866960384224
// MI455X (gfx1250) — matrix operand order repaired and checked by reading, not compiled, not run on hardware
//
#include <hip/hip_runtime.h>
#include <hip/hip_bf16.h>

#define NN 1024
#define TT 128
#define DD 512
#define KK 1024  // concat K = 2*D

typedef __attribute__((ext_vector_type(16))) __bf16 v16bf;
typedef __attribute__((ext_vector_type(8)))  float  v8f;
typedef __attribute__((ext_vector_type(4)))  unsigned int v4u;
typedef __attribute__((ext_vector_type(4)))  int v4i;
typedef __attribute__((ext_vector_type(8)))  int v8i;

union FragBF { v16bf v; uint4 u[2]; };

// LDS layout: rows padded to 80 B (20 dwords).
// B slab: 256 rows (4 gates x 64 cols) x 64 B data  -> 20480 B / buffer
// A slab:  64 rows                     x 64 B data  ->  5120 B / buffer
#define B_SLAB 20480
#define A_SLAB 5120
#define SMEM_BYTES (2 * B_SLAB + 2 * A_SLAB)   // 51200

#if defined(__has_builtin)
#  if __has_builtin(__builtin_amdgcn_sched_barrier)
#    define SCHED_FENCE() __builtin_amdgcn_sched_barrier(0)
#  endif
#endif
#ifndef SCHED_FENCE
#  define SCHED_FENCE() asm volatile("" ::: "memory")
#endif

// ---------------------------------------------------------------------------
// fp32 weights -> bf16, concatenated per gate: Wbuf[g][d][k], k<512 from W*x,
// k>=512 from W*h.  Row-major over k.
// ---------------------------------------------------------------------------
__global__ __launch_bounds__(256) void convert_weights(
    const float* __restrict__ Wgx, const float* __restrict__ Wgh,
    const float* __restrict__ Wix, const float* __restrict__ Wih,
    const float* __restrict__ Wfx, const float* __restrict__ Wfh,
    const float* __restrict__ Wox, const float* __restrict__ Woh,
    __hip_bfloat16* __restrict__ Wbuf)
{
    int idx  = blockIdx.x * blockDim.x + threadIdx.x;   // 0 .. 4*512*1024-1
    int gate = idx >> 19;
    int rem  = idx & ((1 << 19) - 1);
    int d    = rem >> 10;
    int k    = rem & 1023;
    bool isX = (k < DD);
    int  kk  = isX ? k : (k - DD);
    const float* src =
        (gate == 0) ? (isX ? Wgx : Wgh) :
        (gate == 1) ? (isX ? Wix : Wih) :
        (gate == 2) ? (isX ? Wfx : Wfh) :
                      (isX ? Wox : Woh);
    Wbuf[idx] = __float2bfloat16(src[d * DD + kk]);
}

__global__ __launch_bounds__(256) void init_state(
    float* __restrict__ S, __hip_bfloat16* __restrict__ H0)
{
    int idx = blockIdx.x * blockDim.x + threadIdx.x;    // 0 .. 1024*512-1
    S[idx]  = 0.0f;
    H0[idx] = __float2bfloat16(0.0f);
}

__global__ __launch_bounds__(256) void convert_x(
    const float* __restrict__ X, __hip_bfloat16* __restrict__ Xbuf, int t)
{
    int idx = blockIdx.x * blockDim.x + threadIdx.x;    // 0 .. 1024*512-1
    int n = idx >> 9;
    int d = idx & (DD - 1);
    Xbuf[idx] = __float2bfloat16(X[(size_t)n * TT * DD + (size_t)t * DD + d]);
}

// ---------------------------------------------------------------------------
// TDM helper: issue TENSOR_LOAD_TO_LDS with descriptor built per
// CDNA5 ISA.  g1 carries dims/strides/padding,
// group0 carries addresses, group2 carries tensor_dim2 (3D z-extent).
// ---------------------------------------------------------------------------
__device__ __forceinline__ void tdm_load(unsigned lds_addr, const void* gptr,
                                         v8i g1, int tdim2)
{
    unsigned long long ga = (unsigned long long)(uintptr_t)gptr;
    v4u g0;
    g0.x = 1u;                                        // count=1 (valid D#)
    g0.y = lds_addr;                                  // lds_addr [63:32]
    g0.z = (unsigned)(ga & 0xffffffffu);              // global_addr [95:64]
    g0.w = (unsigned)((ga >> 32) & 0x01ffffffu)       // global_addr [120:96]
         | (2u << 30);                                // type=2 (image)
    v4i g2; g2.x = tdim2; g2.y = 0; g2.z = 0; g2.w = 0;
    v4i g3; g3.x = 0; g3.y = 0; g3.z = 0; g3.w = 0;
#if __clang_major__ >= 23
    v8i gz; gz[0]=0; gz[1]=0; gz[2]=0; gz[3]=0; gz[4]=0; gz[5]=0; gz[6]=0; gz[7]=0;
    __builtin_amdgcn_tensor_load_to_lds(g0, g1, g2, g3, gz, 0);
#else
    __builtin_amdgcn_tensor_load_to_lds(g0, g1, g2, g3, 0);
#endif
}

__device__ __forceinline__ unsigned lds_off(const void* p)
{
    // Generic LDS pointer = {SHARED_BASE hi, byte offset}; low 32 bits are the
    // workgroup-relative LDS byte address the D# expects.
    return (unsigned)(uintptr_t)p;
}

// ---------------------------------------------------------------------------
// Fused LSTM step.  Block = 8 waves (4 M x 2 N), block tile 64 rows x 64 cols,
// all 4 gates.  K = 1024 (x-part then h-part), chunked by 32 with TDM
// double-buffering into LDS.  Grid (16, 8).
// ---------------------------------------------------------------------------
__global__ __launch_bounds__(256) void lstm_step(
    const __hip_bfloat16* __restrict__ Xbuf,   // [N, D] bf16 (x_t)
    const __hip_bfloat16* __restrict__ Hin,    // [N, D] bf16 (h_{t-1})
    const __hip_bfloat16* __restrict__ Wbuf,   // [4, D, 2D] bf16
    float* __restrict__ S,                     // [N, D] fp32 cell state
    const float* __restrict__ bg, const float* __restrict__ bi,
    const float* __restrict__ bfv, const float* __restrict__ bo,
    float* __restrict__ out,                   // [N, T, D] fp32
    __hip_bfloat16* __restrict__ Hout,         // [N, D] bf16 (h_t)
    int t)
{
    __shared__ __align__(128) char smem[SMEM_BYTES];

    const int lane  = threadIdx.x & 31;
    const int wave  = threadIdx.x >> 5;
    const int wm    = wave >> 1;          // 0..3  (M)
    const int wn    = wave & 1;           // 0..1  (N)
    const int row0  = blockIdx.x * 64 + wm * 16;
    const int col0  = blockIdx.y * 64 + wn * 32;
    const int l15   = lane & 15;
    const int khalf = lane >> 4;          // 0 or 1

    // Descriptor group1 constants (see §8.4):
    //  [0] data_size=2B | pad_enable | pad_interval=16dw | pad_amount=4dw
    //  [1..3] tensor_dim0 / tensor_dim1 / tile_dim0=32
    //  [4] tile_dim1=64 | tile_dim2       [5..7] dim strides
    v8i g1B, g1A;
    g1B[0] = 0x06D10000; g1B[1] = 0x04000000; g1B[2] = 0x08000000;
    g1B[3] = 0x00200000; g1B[4] = 0x00040040; g1B[5] = 1024;
    g1B[6] = 0;          g1B[7] = 8;          // dim1_stride = 512*1024
    g1A[0] = 0x06D10000; g1A[1] = 0x02000000; g1A[2] = 0x04000000;
    g1A[3] = 0x00200000; g1A[4] = 0x00010040; g1A[5] = 512;
    g1A[6] = 0;          g1A[7] = 0;

    // Buffer byte offsets inside smem (no LDS-pointer arrays: keep all LDS
    // addressing as runtime smem+offset arithmetic so no addrspacecast ever
    // lands in a static initializer).
    const unsigned sbase = lds_off(smem);

    const __hip_bfloat16* const wbase = Wbuf + (size_t)(blockIdx.y * 64) * KK;
    const int arowblk = blockIdx.x * 64;

    v8f acc[4][2] = {};                   // [gate][ntile] fp32 accumulators

    // Per-lane LDS read offsets (loop-invariant except buffer ping-pong).
    const int a_off = 2 * B_SLAB + (wm * 16 + l15) * 80 + khalf * 32;
    const int b_off = (wn * 32 + l15) * 80 + khalf * 32;

    // ---- prologue: issue chunk 0 ----
    if (wave == 0) {
        tdm_load(sbase, wbase, g1B, 4);
        tdm_load(sbase + 2 * B_SLAB, Xbuf + (size_t)arowblk * DD, g1A, 1);
    }

    for (int c = 0; c < 32; ++c) {
        const int bufB = (c & 1) * B_SLAB;
        const int bufA = (c & 1) * A_SLAB;
        if (wave == 0) {
            if (c < 31) {
                const int cn   = c + 1;
                const int part = cn >> 4;          // 0: x-part, 1: h-part
                const int kk   = (cn & 15) * 32;
                const __hip_bfloat16* abase = (part ? Hin : Xbuf)
                                            + (size_t)arowblk * DD + kk;
                tdm_load(sbase + (cn & 1) * B_SLAB,
                         wbase + part * DD + kk, g1B, 4);
                tdm_load(sbase + 2 * B_SLAB + (cn & 1) * A_SLAB,
                         abase, g1A, 1);
                __builtin_amdgcn_s_wait_tensorcnt((short)2); // chunk c done
            } else {
                __builtin_amdgcn_s_wait_tensorcnt((short)0);
            }
        }
        __syncthreads();                  // publish chunk c LDS to all waves

        // Issue ALL fragment loads back-to-back (distinct registers), fence
        // the scheduler, then run the 8 independent WMMAs.  The dscnt waits
        // become partial (14,12,...) so LDS latency overlaps the bursts
        // instead of a per-WMMA s_wait_dscnt 0.
        FragBF a, b[4][2];
        {
            const char* ap = smem + bufA + a_off;
            a.u[0] = *(const uint4*)(ap);
            a.u[1] = *(const uint4*)(ap + 16);
        }
#pragma unroll
        for (int g = 0; g < 4; ++g) {
#pragma unroll
            for (int nt = 0; nt < 2; ++nt) {
                const char* bp = smem + bufB + (g * 64 + nt * 16) * 80 + b_off;
                b[g][nt].u[0] = *(const uint4*)(bp);
                b[g][nt].u[1] = *(const uint4*)(bp + 16);
            }
        }
        SCHED_FENCE();                    // loads above, WMMAs below
#pragma unroll
        for (int g = 0; g < 4; ++g) {
#pragma unroll
            for (int nt = 0; nt < 2; ++nt) {
                acc[g][nt] = __builtin_amdgcn_wmma_f32_16x16x32_bf16(
                    false, a.v, false, b[g][nt].v, (short)0, acc[g][nt],
                    false, false);
            }
        }
        __syncthreads();                  // done reading buf before reuse
    }

    // Epilogue: C layout -> VGPR r, lanes 0-15: M=r, N=lane;
    //                              lanes 16-31: M=r+8, N=lane-16.
#pragma unroll
    for (int nt = 0; nt < 2; ++nt) {
        const int gcol = col0 + nt * 16 + l15;
        const float vbg = bg[gcol], vbi = bi[gcol];
        const float vbf = bfv[gcol], vbo = bo[gcol];
#pragma unroll
        for (int r = 0; r < 8; ++r) {
            const int grow = row0 + khalf * 8 + r;
            const float gg = tanhf(acc[0][nt][r] + vbg);
            const float ii = 1.0f / (1.0f + __expf(-(acc[1][nt][r] + vbi)));
            const float ff = 1.0f / (1.0f + __expf(-(acc[2][nt][r] + vbf)));
            const float oo = 1.0f / (1.0f + __expf(-(acc[3][nt][r] + vbo)));
            const size_t sidx = (size_t)grow * DD + gcol;
            const float s_new = gg * ii + S[sidx] * ff;
            S[sidx] = s_new;
            const float h = s_new * oo;    // reference: no tanh on cell
            out[(size_t)grow * TT * DD + (size_t)t * DD + gcol] = h;
            Hout[sidx] = __float2bfloat16(h);
        }
    }
}

// ---------------------------------------------------------------------------
extern "C" void kernel_launch(void* const* d_in, const int* in_sizes, int n_in,
                              void* d_out, int out_size, void* d_ws, size_t ws_size,
                              hipStream_t stream)
{
    const float* X   = (const float*)d_in[0];
    const float* Wgx = (const float*)d_in[1];
    const float* Wgh = (const float*)d_in[2];
    const float* Wix = (const float*)d_in[3];
    const float* Wih = (const float*)d_in[4];
    const float* Wfx = (const float*)d_in[5];
    const float* Wfh = (const float*)d_in[6];
    const float* Wox = (const float*)d_in[7];
    const float* Woh = (const float*)d_in[8];
    const float* bg  = (const float*)d_in[9];
    const float* bi  = (const float*)d_in[10];
    const float* bfv = (const float*)d_in[11];
    const float* bo  = (const float*)d_in[12];
    float* out = (float*)d_out;

    // Workspace (~9 MiB): Wbuf 4 MiB | Xbuf 1 MiB | H0 1 MiB | H1 1 MiB | S 2 MiB
    char* ws = (char*)d_ws;
    __hip_bfloat16* Wbuf = (__hip_bfloat16*)ws;
    __hip_bfloat16* Xbuf = (__hip_bfloat16*)(ws + (size_t)4 * DD * KK * 2);
    __hip_bfloat16* H0   = Xbuf + (size_t)NN * DD;
    __hip_bfloat16* H1   = H0   + (size_t)NN * DD;
    float*          S    = (float*)(H1 + (size_t)NN * DD);

    convert_weights<<<(4 * DD * KK) / 256, 256, 0, stream>>>(
        Wgx, Wgh, Wix, Wih, Wfx, Wfh, Wox, Woh, Wbuf);
    init_state<<<(NN * DD) / 256, 256, 0, stream>>>(S, H0);

    dim3 grid(NN / 64, DD / 64);   // (16, 8)
    for (int t = 0; t < TT; ++t) {
        convert_x<<<(NN * DD) / 256, 256, 0, stream>>>(X, Xbuf, t);
        const __hip_bfloat16* Hin  = (t & 1) ? H1 : H0;
        __hip_bfloat16*       Hout = (t & 1) ? H0 : H1;
        lstm_step<<<grid, 256, 0, stream>>>(
            Xbuf, Hin, Wbuf, S, bg, bi, bfv, bo, out, Hout, t);
    }
}
